// CHRC_47562467836574
// MI455X (gfx1250) — matrix operand order repaired and checked by reading, not compiled, not run on hardware
//
#include <hip/hip_runtime.h>
#include <hip/hip_bf16.h>

// Problem constants (match the reference).
#define BQ 1024
#define NK 100000
#define DD 512
#define HH 96
#define FF 7
#define KK 16
#define SS 16            // N-dimension splits
#define CH 6250          // keys per chunk (NK / SS, exact)
#define NEG_INF (-3.402823466e38f)

typedef unsigned short ushort_t;
typedef unsigned int   v4u  __attribute__((ext_vector_type(4)));
typedef __bf16         v16bf __attribute__((ext_vector_type(16)));
typedef float          v8f  __attribute__((ext_vector_type(8)));

union V32B { v4u q[2]; v16bf v; };

static __device__ __forceinline__ ushort_t f2bf(float f) {
    unsigned int u = __float_as_uint(f);
    unsigned int r = u + 0x7FFFu + ((u >> 16) & 1u);   // round-to-nearest-even
    return (ushort_t)(r >> 16);
}

// Sorted (descending) 16-slot insert, fully unrolled -> stays in VGPRs.
static __device__ __forceinline__ void topk_insert(float (&ts)[KK], int (&ti)[KK],
                                                   float v, int n) {
#pragma unroll
    for (int j = KK - 1; j >= 0; --j) {
        if (v > ts[j]) {
            if (j == 0 || !(v > ts[j - 1])) { ts[j] = v; ti[j] = n; }
            else                            { ts[j] = ts[j - 1]; ti[j] = ti[j - 1]; }
        }
    }
}

// ---------------------------------------------------------------------------
// Kernel 1: fold 1/||q|| into query, decay/||k|| into keys, convert to bf16.
// rows [0, BQ) = query rows; rows [BQ, BQ+NK) = key rows. 128 thr, 4 elem/thr.
// ---------------------------------------------------------------------------
__global__ __launch_bounds__(128) void CHRC_prep(const float* __restrict__ q,
                                                 const float* __restrict__ k,
                                                 const int* __restrict__ tstamp,
                                                 const int* __restrict__ gstep,
                                                 ushort_t* __restrict__ qn,
                                                 ushort_t* __restrict__ kb) {
    const int row = blockIdx.x;
    const int tid = threadIdx.x;
    const bool isq = row < BQ;
    const float* src = isq ? (q + (size_t)row * DD)
                           : (k + (size_t)(row - BQ) * DD);
    float x[4]; float ss = 0.f;
#pragma unroll
    for (int j = 0; j < 4; ++j) { x[j] = src[tid + 128 * j]; ss += x[j] * x[j]; }
#pragma unroll
    for (int off = 16; off >= 1; off >>= 1) ss += __shfl_xor(ss, off, 32);
    __shared__ float red[4];
    __shared__ float stot;
    if ((tid & 31) == 0) red[tid >> 5] = ss;
    __syncthreads();
    if (tid == 0) stot = red[0] + red[1] + red[2] + red[3];
    __syncthreads();
    float scale = 1.0f / fmaxf(sqrtf(stot), 1e-12f);
    ushort_t* dst;
    if (isq) {
        dst = qn + (size_t)row * DD;
    } else {
        const int n = row - BQ;
        const float age = (float)(gstep[0] - tstamp[n]);
        scale *= __powf(0.995f, age);                  // DECAY_FACTOR^age
        dst = kb + (size_t)n * DD;
    }
#pragma unroll
    for (int j = 0; j < 4; ++j) dst[tid + 128 * j] = f2bf(x[j] * scale);
}

// ---------------------------------------------------------------------------
// Kernel 2: fused bf16-WMMA similarity + chunked top-16.
// grid (64 query-row tiles, 16 key chunks), 128 threads = 4 waves.
// Each wave owns PAIRS of 16-key tiles (16x32 output, 2 accumulators),
// K=512 as 16 double-buffered steps of wmma_f32_16x16x32_bf16 (x2 tiles).
// Double buffering makes the pre-WMMA waits partial (loads return in order),
// overlapping L2/LDS latency with matrix math.
// ---------------------------------------------------------------------------
__global__ __launch_bounds__(128) void CHRC_simtopk(const ushort_t* __restrict__ qn,
                                                    const ushort_t* __restrict__ kb,
                                                    float* __restrict__ psim,
                                                    int* __restrict__ pidx) {
    __shared__ __align__(16) ushort_t A[16 * 520];     // padded A tile (bf16 bits)
    __shared__ float stg[4][16][17];                   // per-wave 16x16 transpose stage
    __shared__ float cS[16][128];                      // merge candidates (sims)
    __shared__ int   cI[16][128];                      // merge candidates (indices)

    const int btile = blockIdx.x;
    const int s     = blockIdx.y;
    const int tid   = threadIdx.x;
    const int w     = tid >> 5;
    const int lane  = tid & 31;
    const int r     = lane & 15;       // matrix row (A) / key column (B) within tile
    const int hi    = lane >> 4;       // lane half

    // Stage the 16x512 bf16 query tile into padded LDS.
    for (int e = tid; e < 16 * DD; e += 128) {
        const int rr = e >> 9, kk2 = e & 511;
        A[rr * 520 + kk2] = qn[((size_t)(btile * 16 + rr)) * DD + kk2];
    }
    __syncthreads();

    float tsv[KK]; int tiv[KK];
#pragma unroll
    for (int j = 0; j < KK; ++j) { tsv[j] = NEG_INF; tiv[j] = 0; }

    const int ns = s * CH;
    const int ne = ns + CH;
    const int T  = (CH + 15) >> 4;                     // 391 tiles per chunk
    const int P  = (T + 1) >> 1;                       // 196 tile-pairs

    const ushort_t* arow = &A[r * 520];

    for (int p = w; p < P; p += 4) {
        const int t0 = p << 1;
        const int ntile0 = ns + t0 * 16;
        const int ntile1 = ntile0 + 16;
        int nc0 = ntile0 + r; nc0 = nc0 < (NK - 1) ? nc0 : (NK - 1);
        int nc1 = ntile1 + r; nc1 = nc1 < (NK - 1) ? nc1 : (NK - 1);
        const ushort_t* kr0 = kb + (size_t)nc0 * DD;
        const ushort_t* kr1 = kb + (size_t)nc1 * DD;

        V32B a[2], b0[2], b1[2];
        // Prologue: operands for K-step 0.
        {
            const int k0 = (hi << 3);
            a[0].q[0] = *(const v4u*)(arow + k0);
            a[0].q[1] = *(const v4u*)(arow + k0 + 16);
            const v4u* bp0 = (const v4u*)(kr0 + (hi << 3));
            b0[0].q[0] = bp0[0]; b0[0].q[1] = bp0[2];
            const v4u* bp1 = (const v4u*)(kr1 + (hi << 3));
            b1[0].q[0] = bp1[0]; b1[0].q[1] = bp1[2];
        }
        v8f c0 = {}, c1 = {};
#pragma unroll
        for (int st = 0; st < 16; ++st) {
            const int cur = st & 1, nxt = cur ^ 1;
            if (st < 15) {                             // prefetch next K-step
                const int ks = (st + 1) << 5;
                const int k0 = ks + (hi << 3);
                a[nxt].q[0] = *(const v4u*)(arow + k0);
                a[nxt].q[1] = *(const v4u*)(arow + k0 + 16);
                const v4u* bp0 = (const v4u*)(kr0 + ks + (hi << 3));
                b0[nxt].q[0] = bp0[0]; b0[nxt].q[1] = bp0[2];
                const v4u* bp1 = (const v4u*)(kr1 + ks + (hi << 3));
                b1[nxt].q[0] = bp1[0]; b1[nxt].q[1] = bp1[2];
            }
            c0 = __builtin_amdgcn_wmma_f32_16x16x32_bf16(
                     false, a[cur].v, false, b0[cur].v, (short)0, c0, false, false);
            c1 = __builtin_amdgcn_wmma_f32_16x16x32_bf16(
                     false, a[cur].v, false, b1[cur].v, (short)0, c1, false, false);
        }

        // Transpose + scan tile 0, then tile 1 (per-wave LDS stage, in-order DS).
#pragma unroll
        for (int v = 0; v < 8; ++v) stg[w][v + (hi << 3)][r] = c0[v];
        __builtin_amdgcn_wave_barrier();
#pragma unroll
        for (int j = 0; j < 8; ++j) {
            const int col = (hi << 3) + j;
            const int n = ntile0 + col;
            const float sv = stg[w][r][col];
            if (n < ne && sv > tsv[KK - 1]) topk_insert(tsv, tiv, sv, n);
        }
        __builtin_amdgcn_wave_barrier();
#pragma unroll
        for (int v = 0; v < 8; ++v) stg[w][v + (hi << 3)][r] = c1[v];
        __builtin_amdgcn_wave_barrier();
#pragma unroll
        for (int j = 0; j < 8; ++j) {
            const int col = (hi << 3) + j;
            const int n = ntile1 + col;
            const float sv = stg[w][r][col];
            if (n < ne && sv > tsv[KK - 1]) topk_insert(tsv, tiv, sv, n);
        }
        __builtin_amdgcn_wave_barrier();
    }

    // Merge: 4 waves x 2 lane-halves = 8 partial lists of 16 per row.
    const int cb = (((w << 1) + hi) << 4);
#pragma unroll
    for (int j = 0; j < KK; ++j) { cS[r][cb + j] = tsv[j]; cI[r][cb + j] = tiv[j]; }
    __syncthreads();

    if (tid < 16) {
        float fs[KK]; int fi[KK];
#pragma unroll
        for (int j = 0; j < KK; ++j) { fs[j] = NEG_INF; fi[j] = 0; }
        for (int m = 0; m < 128; ++m) {
            const float v = cS[tid][m];
            if (v > fs[KK - 1]) topk_insert(fs, fi, v, cI[tid][m]);
        }
        const size_t base = ((size_t)(btile * 16 + tid) * SS + s) * KK;
#pragma unroll
        for (int j = 0; j < KK; ++j) { psim[base + j] = fs[j]; pidx[base + j] = fi[j]; }
    }
}

// ---------------------------------------------------------------------------
// Kernel 3: per-query merge of S*K=256 candidates -> top-16, softmax weights,
// weighted gather-sum of values. 1024 blocks x 256 threads.
// ---------------------------------------------------------------------------
__global__ __launch_bounds__(256) void CHRC_finalize(const float* __restrict__ psim,
                                                     const int* __restrict__ pidx,
                                                     const float* __restrict__ values,
                                                     float* __restrict__ out) {
    const int b = blockIdx.x;
    const int tid = threadIdx.x;
    __shared__ float chS[KK];
    __shared__ int   chI[KK];
    __shared__ float wgt[KK];

    if (tid < 32) {
        float cs[8]; int ci[8];
#pragma unroll
        for (int j = 0; j < 8; ++j) {
            cs[j] = psim[(size_t)b * (SS * KK) + tid * 8 + j];
            ci[j] = pidx[(size_t)b * (SS * KK) + tid * 8 + j];
        }
#pragma unroll 1
        for (int rsel = 0; rsel < KK; ++rsel) {
            float lv = NEG_INF; int ls = 0;
#pragma unroll
            for (int j = 0; j < 8; ++j) if (cs[j] > lv) { lv = cs[j]; ls = j; }
            float gm = lv;
#pragma unroll
            for (int off = 16; off >= 1; off >>= 1)
                gm = fmaxf(gm, __shfl_xor(gm, off, 32));
            const unsigned long long bal = __ballot(lv == gm);
            const int first = __ffsll(bal) - 1;
            if (tid == first) { chS[rsel] = gm; chI[rsel] = ci[ls]; cs[ls] = NEG_INF; }
        }
    }
    __syncthreads();
    if (tid == 0) {
        // softmax(top/T) -> mask (>= MIN_SIMILARITY) -> renormalize (+EPS)
        const float mx = chS[0];
        float e[KK]; float se = 0.f;
#pragma unroll
        for (int j = 0; j < KK; ++j) { e[j] = __expf((chS[j] - mx) * 10.0f); se += e[j]; }
        float s2 = 0.f;
#pragma unroll
        for (int j = 0; j < KK; ++j) {
            float wv = e[j] / se;
            wv *= (chS[j] >= 0.0f) ? 1.0f : 0.0f;
            wgt[j] = wv; s2 += wv;
        }
        s2 += 1e-8f;
#pragma unroll
        for (int j = 0; j < KK; ++j) wgt[j] /= s2;
    }
    __syncthreads();
    for (int e2 = tid; e2 < HH * FF; e2 += 256) {
        float acc = 0.f;
#pragma unroll
        for (int j = 0; j < KK; ++j)
            acc += wgt[j] * values[(size_t)chI[j] * (HH * FF) + e2];
        out[(size_t)b * (HH * FF) + e2] = acc;
    }
}

// ---------------------------------------------------------------------------
extern "C" void kernel_launch(void* const* d_in, const int* in_sizes, int n_in,
                              void* d_out, int out_size, void* d_ws, size_t ws_size,
                              hipStream_t stream) {
    (void)in_sizes; (void)n_in; (void)out_size; (void)ws_size;
    const float* query  = (const float*)d_in[0];
    const float* keys   = (const float*)d_in[1];
    const float* values = (const float*)d_in[2];
    const int*   tstamp = (const int*)d_in[3];
    const int*   gstep  = (const int*)d_in[4];
    // d_in[5] = top_k (fixed at 16 here)
    float* out = (float*)d_out;

    // Workspace layout.
    char* ws = (char*)d_ws;
    ushort_t* qn   = (ushort_t*)(ws);                                   // 1,048,576 B
    ushort_t* kbuf = (ushort_t*)(ws + (size_t)BQ * DD * 2);             // 102,400,000 B
    float*    psim = (float*)(ws + (size_t)BQ * DD * 2 + (size_t)NK * DD * 2);
    int*      pidx = (int*)((char*)psim + (size_t)BQ * SS * KK * 4);

    CHRC_prep<<<BQ + NK, 128, 0, stream>>>(query, keys, tstamp, gstep, qn, kbuf);
    CHRC_simtopk<<<dim3(BQ / 16, SS), 128, 0, stream>>>(qn, kbuf, psim, pidx);
    CHRC_finalize<<<BQ, 256, 0, stream>>>(psim, pidx, values, out);
}
